// MultiHeadAttention_78666620994248
// MI455X (gfx1250) — matrix operand order repaired and checked by reading, not compiled, not run on hardware
//
#include <hip/hip_runtime.h>
#include <hip/hip_bf16.h>
#include <math.h>

// ---------------------------------------------------------------------------
// MI455X (gfx1250) fused multi-head attention, bf16 WMMA + Tensor Data Mover.
//   B=4, H=16, T=2048, E=1024, head dim S=64.
// GEMMs: v_wmma_f32_16x16x32_bf16 (f32 accum). Attention K/V tiles are staged
// into LDS by TDM (tensor_load_to_lds) with double buffering when available.
// ---------------------------------------------------------------------------

#define BATCH 4
#define HEADS 16
#define SEQ   2048
#define EMB   1024
#define HD    64

typedef __attribute__((ext_vector_type(16))) __bf16 v16bf;
typedef __attribute__((ext_vector_type(8)))  float  v8f;
typedef unsigned int u32x4 __attribute__((ext_vector_type(4)));
typedef int          i32x8 __attribute__((ext_vector_type(8)));
typedef int          i32x4 __attribute__((ext_vector_type(4)));

#if __has_builtin(__builtin_amdgcn_tensor_load_to_lds)
#define ATHENA_TDM 1
#else
#define ATHENA_TDM 0
#endif
#if __has_include(<hip/amd_detail/amd_gfx1250_TDM.h>)
#define ATHENA_TDM_6ARG 1
#else
#define ATHENA_TDM_6ARG 0
#endif

union AFragU { v16bf v; unsigned int u[8]; };

// f32 -> bf16, round-to-nearest-even
__device__ __forceinline__ unsigned short f32_to_bf16(float f) {
  unsigned int u = __float_as_uint(f);
  unsigned int r = u + 0x7FFFu + ((u >> 16) & 1u);
  return (unsigned short)(r >> 16);
}
__device__ __forceinline__ unsigned int pack_bf16x2(float lo, float hi) {
  return (unsigned int)f32_to_bf16(lo) | ((unsigned int)f32_to_bf16(hi) << 16);
}

// A fragment: 16(M)x32(K) bf16.  Lane l holds row M=l%16.
// Pair (2r,2r+1) at K = (r/4)*16 + (l/16)*8 + (r%4)*2  (ISA 7.12.2).
__device__ __forceinline__ v16bf lds_afrag(const unsigned short* base, int stride,
                                           int k0, int lane) {
  const int half = lane >> 4, ln = lane & 15;
  AFragU f;
  const unsigned short* rp = base + ln * stride + k0 + half * 8;
#pragma unroll
  for (int r = 0; r < 8; ++r)
    f.u[r] = *(const unsigned int*)(rp + (r >> 2) * 16 + ((r & 3) << 1));
  return f.v;
}

// B fragment: 32(K)x16(N) bf16, LDS stored "n-major": addr = n*stride + k.
__device__ __forceinline__ v16bf lds_bfrag(const unsigned short* base, int stride,
                                           int k0, int lane) {
  const int half = lane >> 4, ln = lane & 15;
  AFragU f;
  const unsigned short* cp = base + ln * stride + k0 + half * 8;
#pragma unroll
  for (int r = 0; r < 8; ++r)
    f.u[r] = *(const unsigned int*)(cp + (r >> 2) * 16 + ((r & 3) << 1));
  return f.v;
}

__device__ __forceinline__ float rmax16(float v) {
#pragma unroll
  for (int m = 1; m < 16; m <<= 1) v = fmaxf(v, __shfl_xor(v, m, 32));
  return v;
}
__device__ __forceinline__ float rsum16(float v) {
#pragma unroll
  for (int m = 1; m < 16; m <<= 1) v += __shfl_xor(v, m, 32);
  return v;
}

#if ATHENA_TDM
// Issue a 2D TDM tile load (bf16 elements) into LDS with row padding that
// produces a 72-half (144B) LDS row stride for 64-half tile rows.
// Descriptor packing per CDNA5 ISA 8.3/8.4.
__device__ __forceinline__ void tdm_load_tile_2d(
    const unsigned short* gsrc, unsigned lds_byte_off,
    unsigned tensor_d0, unsigned tensor_d1, unsigned d0_stride,
    unsigned tile_d0, unsigned tile_d1) {
  const unsigned long long ga = (unsigned long long)(uintptr_t)gsrc;
  u32x4 g0;
  g0[0] = 1u;                                                  // count=1
  g0[1] = lds_byte_off;                                        // lds_addr
  g0[2] = (unsigned)ga;                                        // global_addr lo
  g0[3] = (unsigned)((ga >> 32) & 0x01ffffffu) | (2u << 30);   // addr hi | type=2
  i32x8 g1;
  g1[0] = (int)((1u << 16)      // data_size = 2 bytes
              | (1u << 20)      // pad_enable
              | (4u << 22)      // pad_interval: 32 dwords (=64 halfs)
              | (3u << 25));    // pad_amount: 4 dwords (16B) -> 72-half stride
  g1[1] = (int)((tensor_d0 & 0xffffu) << 16);                        // dim0 lo
  g1[2] = (int)((tensor_d0 >> 16) | ((tensor_d1 & 0xffffu) << 16));  // dim0 hi|dim1 lo
  g1[3] = (int)((tensor_d1 >> 16) | ((tile_d0 & 0xffffu) << 16));    // dim1 hi|tile0
  g1[4] = (int)(tile_d1 & 0xffffu);                                  // tile1, tile2=0
  g1[5] = (int)d0_stride;                                            // dim0 stride lo
  g1[6] = 0;                                                         // stride hi, dim1_stride=0
  g1[7] = 0;
  i32x4 g2; g2[0] = 1; g2[1] = 0; g2[2] = 0; g2[3] = 0;  // tensor_dim2=1 (unused)
  i32x4 g3; g3[0] = 0; g3[1] = 0; g3[2] = 0; g3[3] = 0;
#if ATHENA_TDM_6ARG
  i32x8 g4;
#pragma unroll
  for (int i = 0; i < 8; ++i) g4[i] = 0;
  __builtin_amdgcn_tensor_load_to_lds(g0, g1, g2, g3, g4, 0);
#else
  __builtin_amdgcn_tensor_load_to_lds(g0, g1, g2, g3, 0);
#endif
}
__device__ __forceinline__ unsigned lds_byte_off(const void* p) {
  return (unsigned)(uintptr_t)p;
}
#endif

// ---------------------------------------------------------------------------
// Kernel 1: Y = x @ W.T (f32 in, bf16 out), head layout. transpose_out:
//   0 -> outh[b][h][t][sd]   (Q, K)
//   1 -> outh[b][h][sd][t]   (V, pre-transposed for the PV GEMM / TDM)
// ---------------------------------------------------------------------------
#define ASTR 40
#define BSTR 40

__global__ __launch_bounds__(256) void qkv_proj_kernel(
    const float* __restrict__ x, const float* __restrict__ W,
    unsigned short* __restrict__ outh, const int transpose_out) {
  const int nb = blockIdx.x * 64;
  const int mb = blockIdx.y * 128;
  const int tid = threadIdx.x;
  const int wave = tid >> 5, lane = tid & 31;
  const int half = lane >> 4, ln = lane & 15;

  __shared__ __align__(16) unsigned short la[128 * ASTR];
  __shared__ __align__(16) unsigned short lb[64 * BSTR];

  v8f z = {};
  v8f acc[4] = {z, z, z, z};

  for (int kk = 0; kk < EMB / 32; ++kk) {
    __syncthreads();
    {  // stage A: x[mb..mb+127][kk*32..+32) -> bf16
      const int row = tid >> 1, cb = (tid & 1) * 16;
      const float4* src = (const float4*)(x + (size_t)(mb + row) * EMB + kk * 32 + cb);
      unsigned short* dst = &la[row * ASTR + cb];
#pragma unroll
      for (int j = 0; j < 4; ++j) {
        float4 v = src[j];
        *(unsigned int*)(dst + 4 * j)     = pack_bf16x2(v.x, v.y);
        *(unsigned int*)(dst + 4 * j + 2) = pack_bf16x2(v.z, v.w);
      }
    }
    {  // stage B: W[nb+n][k] -> lb[n][k] (n-major)
      const int n = tid >> 2, cb = (tid & 3) * 8;
      const float4* src = (const float4*)(W + (size_t)(nb + n) * EMB + kk * 32 + cb);
      unsigned short* dst = &lb[n * BSTR + cb];
#pragma unroll
      for (int j = 0; j < 2; ++j) {
        float4 v = src[j];
        *(unsigned int*)(dst + 4 * j)     = pack_bf16x2(v.x, v.y);
        *(unsigned int*)(dst + 4 * j + 2) = pack_bf16x2(v.z, v.w);
      }
    }
    __syncthreads();
    v16bf a = lds_afrag(la + wave * 16 * ASTR, ASTR, 0, lane);
#pragma unroll
    for (int nt = 0; nt < 4; ++nt) {
      v16bf b = lds_bfrag(lb + nt * 16 * BSTR, BSTR, 0, lane);
      acc[nt] = __builtin_amdgcn_wmma_f32_16x16x32_bf16(false, a, false, b,
                                                        (short)0, acc[nt], false, false);
    }
  }

#pragma unroll
  for (int nt = 0; nt < 4; ++nt) {
#pragma unroll
    for (int r = 0; r < 8; ++r) {
      const int m = mb + wave * 16 + half * 8 + r;
      const int n = nb + nt * 16 + ln;
      const int bb = m >> 11, t = m & (SEQ - 1);
      const int h = n >> 6, sd = n & (HD - 1);
      const size_t head = (size_t)(bb * HEADS + h);
      const size_t idx = transpose_out ? (head * HD + sd) * SEQ + t
                                       : (head * SEQ + t) * HD + sd;
      outh[idx] = f32_to_bf16(acc[nt][r]);
    }
  }
}

// ---------------------------------------------------------------------------
// Kernel 2: fused flash-style attention per (b,h), 128 query rows per block.
//   S = Q K^T ; scaled = floor(S/32) ; online softmax ; O += P V ; O /= l.
// K/V tiles staged by TDM with double buffering (fallback: manual copies).
// ---------------------------------------------------------------------------
#define QSTR 72
#define KSTR 72
#define PSTR 72
#define NTILE (SEQ / 64)

__global__ __launch_bounds__(256) void attn_fused_kernel(
    const unsigned short* __restrict__ qh, const unsigned short* __restrict__ kh,
    const unsigned short* __restrict__ vh, unsigned short* __restrict__ attn_out) {
  const int qt = blockIdx.x;   // 0..15
  const int bh = blockIdx.y;   // 0..63
  const int bb = bh >> 4, h = bh & (HEADS - 1);
  const int tid = threadIdx.x;
  const int wave = tid >> 5, lane = tid & 31;
  const int half = lane >> 4, ln = lane & 15;

  __shared__ __align__(16) unsigned short lq[128 * QSTR];      // Q [qrow][kd]
  __shared__ __align__(16) unsigned short lp[8 * 16 * PSTR];   // P per wave
#if ATHENA_TDM
  __shared__ __align__(16) unsigned short lk[2][64 * KSTR];    // K [key][kd]
  __shared__ __align__(16) unsigned short lv[2][64 * KSTR];    // V [sd][key]
#else
  __shared__ __align__(16) unsigned short lk[1][64 * KSTR];
  __shared__ __align__(16) unsigned short lv[1][64 * KSTR];
#endif

  const unsigned short* qg = qh + ((size_t)bh * SEQ + qt * 128) * HD;  // [t][sd]
  const unsigned short* kg = kh + (size_t)bh * SEQ * HD;               // [t][sd]
  const unsigned short* vg = vh + (size_t)bh * HD * SEQ;               // [sd][t]

#if ATHENA_TDM
  if (wave == 0) {
    // Q: tensor [SEQ x HD], tile [128 rows x 64]
    tdm_load_tile_2d(qg, lds_byte_off(lq), HD, SEQ, HD, HD, 128);
    // tile 0 of K and V
    tdm_load_tile_2d(kg, lds_byte_off(lk[0]), HD, SEQ, HD, HD, 64);
    tdm_load_tile_2d(vg, lds_byte_off(lv[0]), SEQ, HD, SEQ, 64, HD);
  }
#else
  {  // stage Q once (already bf16)
    const int row = tid >> 1, cb = (tid & 1) * 32;
    const uint4* src = (const uint4*)(qg + (size_t)row * HD + cb);
    uint4* dst = (uint4*)&lq[row * QSTR + cb];
#pragma unroll
    for (int j = 0; j < 4; ++j) dst[j] = src[j];
  }
#endif

  v8f z = {};
  v8f oacc[4] = {z, z, z, z};
  float m_i[8], l_i[8];
#pragma unroll
  for (int r = 0; r < 8; ++r) { m_i[r] = -1.0e30f; l_i[r] = 0.0f; }

  for (int kt = 0; kt < NTILE; ++kt) {
    __syncthreads();   // all readers of the buffer we are about to refill done
#if ATHENA_TDM
    if (wave == 0) {
      if (kt + 1 < NTILE) {
        const int nb2 = (kt + 1) & 1;
        tdm_load_tile_2d(kg + (size_t)(kt + 1) * 64 * HD, lds_byte_off(lk[nb2]),
                         HD, SEQ, HD, HD, 64);
        tdm_load_tile_2d(vg + (size_t)(kt + 1) * 64, lds_byte_off(lv[nb2]),
                         SEQ, HD, SEQ, 64, HD);
        __builtin_amdgcn_s_wait_tensorcnt((short)2);  // tile kt complete
      } else {
        __builtin_amdgcn_s_wait_tensorcnt((short)0);
      }
    }
    const unsigned short* lkc = lk[kt & 1];
    const unsigned short* lvc = lv[kt & 1];
#else
    if (kt + 1 < NTILE) {
      __builtin_prefetch(kg + (size_t)(kt + 1) * 64 * HD + tid * 16, 0, 0);
      __builtin_prefetch(vg + (size_t)(kt + 1) * 64 + tid * 16, 0, 0);
    }
    {  // stage K tile [key][kd]
      const int key = tid >> 2, cb = (tid & 3) * 16;
      const uint4* src = (const uint4*)(kg + (size_t)(kt * 64 + key) * HD + cb);
      uint4* dst = (uint4*)&lk[0][key * KSTR + cb];
      dst[0] = src[0];
      dst[1] = src[1];
    }
    {  // stage V tile [sd][key] (global V already transposed)
      const int sd = tid >> 2, cb = (tid & 3) * 16;
      const uint4* src = (const uint4*)(vg + (size_t)sd * SEQ + kt * 64 + cb);
      uint4* dst = (uint4*)&lv[0][sd * KSTR + cb];
      dst[0] = src[0];
      dst[1] = src[1];
    }
    const unsigned short* lkc = lk[0];
    const unsigned short* lvc = lv[0];
#endif
    __syncthreads();   // tile kt (and on first pass, Q) ready

    // ---- S = Q K^T  (M=16 per wave, N=64 keys, K=64 dims) ----
    v8f sacc[4] = {z, z, z, z};
#pragma unroll
    for (int ks = 0; ks < 2; ++ks) {
      v16bf a = lds_afrag(lq + wave * 16 * QSTR, QSTR, ks * 32, lane);
#pragma unroll
      for (int nt = 0; nt < 4; ++nt) {
        v16bf b = lds_bfrag(lkc + nt * 16 * KSTR, KSTR, ks * 32, lane);
        sacc[nt] = __builtin_amdgcn_wmma_f32_16x16x32_bf16(false, a, false, b,
                                                           (short)0, sacc[nt], false, false);
      }
    }
    // reference quirk: floor-divide by sqrt(E)=32
#pragma unroll
    for (int nt = 0; nt < 4; ++nt)
#pragma unroll
      for (int r = 0; r < 8; ++r)
        sacc[nt][r] = floorf(sacc[nt][r] * 0.03125f);

    // ---- online softmax; emit P (bf16) to per-wave LDS ----
    unsigned short* pw = lp + wave * (16 * PSTR);
#pragma unroll
    for (int r = 0; r < 8; ++r) {
      float mc = -1.0e30f;
#pragma unroll
      for (int nt = 0; nt < 4; ++nt) mc = fmaxf(mc, sacc[nt][r]);
      mc = rmax16(mc);
      const float mn = fmaxf(m_i[r], mc);
      const float alpha = __expf(m_i[r] - mn);
      float ps = 0.0f;
#pragma unroll
      for (int nt = 0; nt < 4; ++nt) {
        const float p = __expf(sacc[nt][r] - mn);
        ps += p;
        pw[(half * 8 + r) * PSTR + nt * 16 + ln] = f32_to_bf16(p);
      }
      ps = rsum16(ps);
      l_i[r] = l_i[r] * alpha + ps;
      m_i[r] = mn;
#pragma unroll
      for (int nt = 0; nt < 4; ++nt) oacc[nt][r] *= alpha;
    }
    __syncthreads();   // P visible

    // ---- O += P V  (K=64 keys; V tile stored [sd][key] -> n-major B) ----
#pragma unroll
    for (int ks = 0; ks < 2; ++ks) {
      v16bf a = lds_afrag(pw, PSTR, ks * 32, lane);
#pragma unroll
      for (int nt = 0; nt < 4; ++nt) {
        v16bf b = lds_bfrag(lvc + nt * 16 * KSTR, KSTR, ks * 32, lane);
        oacc[nt] = __builtin_amdgcn_wmma_f32_16x16x32_bf16(false, a, false, b,
                                                           (short)0, oacc[nt], false, false);
      }
    }
  }

  // finalize: O /= l ; write attn in [b, t, e] layout (bf16)
#pragma unroll
  for (int nt = 0; nt < 4; ++nt) {
#pragma unroll
    for (int r = 0; r < 8; ++r) {
      const int t = qt * 128 + wave * 16 + half * 8 + r;
      const int sd = nt * 16 + ln;
      const float o = oacc[nt][r] / l_i[r];
      attn_out[((size_t)(bb * SEQ) + t) * EMB + h * HD + sd] = f32_to_bf16(o);
    }
  }
}

// ---------------------------------------------------------------------------
// Kernel 3: out = attn(bf16) @ Wff.T + bff  (f32 out)
// ---------------------------------------------------------------------------
__global__ __launch_bounds__(256) void out_proj_kernel(
    const unsigned short* __restrict__ attn, const float* __restrict__ W,
    const float* __restrict__ bias, float* __restrict__ out) {
  const int nb = blockIdx.x * 64;
  const int mb = blockIdx.y * 128;
  const int tid = threadIdx.x;
  const int wave = tid >> 5, lane = tid & 31;
  const int half = lane >> 4, ln = lane & 15;

  __shared__ __align__(16) unsigned short la[128 * ASTR];
  __shared__ __align__(16) unsigned short lb[64 * BSTR];

  v8f z = {};
  v8f acc[4] = {z, z, z, z};

  for (int kk = 0; kk < EMB / 32; ++kk) {
    __syncthreads();
    {  // stage A (already bf16)
      const int row = tid >> 1, cb = (tid & 1) * 16;
      const uint4* src = (const uint4*)(attn + (size_t)(mb + row) * EMB + kk * 32 + cb);
      uint4* dst = (uint4*)&la[row * ASTR + cb];
      dst[0] = src[0];
      dst[1] = src[1];
    }
    {  // stage B: Wff[nb+n][k] -> lb[n][k]
      const int n = tid >> 2, cb = (tid & 3) * 8;
      const float4* src = (const float4*)(W + (size_t)(nb + n) * EMB + kk * 32 + cb);
      unsigned short* dst = &lb[n * BSTR + cb];
#pragma unroll
      for (int j = 0; j < 2; ++j) {
        float4 v = src[j];
        *(unsigned int*)(dst + 4 * j)     = pack_bf16x2(v.x, v.y);
        *(unsigned int*)(dst + 4 * j + 2) = pack_bf16x2(v.z, v.w);
      }
    }
    __syncthreads();
    v16bf a = lds_afrag(la + wave * 16 * ASTR, ASTR, 0, lane);
#pragma unroll
    for (int nt = 0; nt < 4; ++nt) {
      v16bf b = lds_bfrag(lb + nt * 16 * BSTR, BSTR, 0, lane);
      acc[nt] = __builtin_amdgcn_wmma_f32_16x16x32_bf16(false, a, false, b,
                                                        (short)0, acc[nt], false, false);
    }
  }

#pragma unroll
  for (int nt = 0; nt < 4; ++nt) {
#pragma unroll
    for (int r = 0; r < 8; ++r) {
      const int m = mb + wave * 16 + half * 8 + r;
      const int n = nb + nt * 16 + ln;
      out[(size_t)m * EMB + n] = acc[nt][r] + bias[n];
    }
  }
}

// ---------------------------------------------------------------------------
extern "C" void kernel_launch(void* const* d_in, const int* in_sizes, int n_in,
                              void* d_out, int out_size, void* d_ws, size_t ws_size,
                              hipStream_t stream) {
  const float* x   = (const float*)d_in[0];
  const float* Wq  = (const float*)d_in[1];
  const float* Wk  = (const float*)d_in[2];
  const float* Wv  = (const float*)d_in[3];
  const float* Wff = (const float*)d_in[4];
  const float* bff = (const float*)d_in[5];
  float* out = (float*)d_out;

  // workspace: 4 bf16 tensors of B*H*T*HD == B*T*E == 8388608 elements (64 MB)
  const size_t HSZ = (size_t)BATCH * HEADS * SEQ * HD;
  unsigned short* qh = (unsigned short*)d_ws;
  unsigned short* kh = qh + HSZ;
  unsigned short* vh = kh + HSZ;   // V stored [b,h,sd,t] (transposed)
  unsigned short* aw = vh + HSZ;

  dim3 blk(256);
  dim3 gproj(EMB / 64, (BATCH * SEQ) / 128);   // 16 x 64
  dim3 gattn(SEQ / 128, BATCH * HEADS);        // 16 x 64

  qkv_proj_kernel<<<gproj, blk, 0, stream>>>(x, Wq, qh, 0);
  qkv_proj_kernel<<<gproj, blk, 0, stream>>>(x, Wk, kh, 0);
  qkv_proj_kernel<<<gproj, blk, 0, stream>>>(x, Wv, vh, 1);
  attn_fused_kernel<<<gattn, blk, 0, stream>>>(qh, kh, vh, aw);
  out_proj_kernel<<<gproj, blk, 0, stream>>>(aw, Wff, bff, out);
}
